// MultivariateTSModel_86346022519289
// MI455X (gfx1250) — matrix operand order repaired and checked by reading, not compiled, not run on hardware
//
#include <hip/hip_runtime.h>
#include <math.h>

// ---------------- model dimensions ----------------
#define Bn   48
#define Tn   200
#define Cn   53
#define En   32
#define Hn   64
#define C2n  2809        // C*C
#define NRn  2544        // B*C rows of GRU state
#define G3n  192         // 3*H
#define E1n  140
#define E2n  1404
#define BTn  9600        // B*T

typedef __attribute__((ext_vector_type(16))) _Float16 v16h;
typedef __attribute__((ext_vector_type(8)))  float    v8f;

#define CVT4(dst, base, f)                 \
    dst[base + 0] = (_Float16)(f).x;       \
    dst[base + 1] = (_Float16)(f).y;       \
    dst[base + 2] = (_Float16)(f).z;       \
    dst[base + 3] = (_Float16)(f).w;

// Tile store epilogue: wave-uniform interior test -> branch-free store path
// (scalar s_cmp/s_cbranch only); boundary tiles fall back to predicated
// stores on a precomputed add-chain address (no per-store 64-bit mul).
__device__ __forceinline__ void store_tile(float* __restrict__ Out, int ldo,
                                           int row0, int col0, int M, int Nn,
                                           int lane, const v8f& acc)
{
    int n    = lane & 15;
    int moff = (lane < 16) ? 0 : 8;
    long base = (long)(row0 + moff) * ldo + col0 + n;
    if (row0 + 16 <= M && col0 + 16 <= Nn) {        // uniform: interior tile
#pragma unroll
        for (int r = 0; r < 8; ++r) Out[base + (long)r * ldo] = acc[r];
    } else {
        bool nok = (col0 + n) < Nn;
#pragma unroll
        for (int r = 0; r < 8; ++r)
            if (nok && (row0 + moff + r) < M) Out[base + (long)r * ldo] = acc[r];
    }
}

// =====================================================================
// K=64 GEMM:  Out[m,n] = sum_{k<64} A[m,k] * W[n,k]   (Out = A @ W^T)
// One wave per 16x16 tile. Branch-free fragment loads:
//  - row/col indices are CLAMPED (garbage rows/cols are never stored),
//  - K==64 exactly -> no k-bounds checks,
//  - lda/ldw multiples of 4 and 16B-aligned bases -> b128 loads.
// Batched via blockIdx.z.
// =====================================================================
__global__ void gemm_k64(const float* __restrict__ A, int lda, long sA,
                         const float* __restrict__ W, int ldw, long sW,
                         float* __restrict__ Out, int ldo, long sO,
                         int M, int Nn)
{
    int lane = threadIdx.x & 31;
    int z = blockIdx.z;
    A   += (long)z * sA;
    W   += (long)z * sW;
    Out += (long)z * sO;

    int row0 = blockIdx.y * 16;
    int col0 = blockIdx.x * 16;
    int mrow = row0 + (lane & 15); if (mrow >= M)  mrow = M - 1;   // clamp, no predication
    int ncol = col0 + (lane & 15); if (ncol >= Nn) ncol = Nn - 1;
    int kbA = (lane < 16) ? 0 : 16;
    int kbB = (lane < 16) ? 0 : 16;
    const float* arow = A + (long)mrow * lda;
    const float* wrow = W + (long)ncol * ldw;

    v8f acc = {};
#pragma unroll
    for (int k0 = 0; k0 < 64; k0 += 32) {
        float4 a0 = *(const float4*)(arow + k0 + kbA);
        float4 a1 = *(const float4*)(arow + k0 + kbA + 4);
        float4 a2 = *(const float4*)(arow + k0 + kbA + 8);
        float4 a3 = *(const float4*)(arow + k0 + kbA + 12);
        float4 b0 = *(const float4*)(wrow + k0 + kbB);
        float4 b1 = *(const float4*)(wrow + k0 + kbB + 4);
        float4 b2 = *(const float4*)(wrow + k0 + kbB + 8);
        float4 b3 = *(const float4*)(wrow + k0 + kbB + 12);
        v16h af, bf;
        CVT4(af, 0, a0)  CVT4(af, 4, a1)  CVT4(af, 8, a2)  CVT4(af, 12, a3)
        CVT4(bf, 0, b0)  CVT4(bf, 4, b1)  CVT4(bf, 8, b2)  CVT4(bf, 12, b3)
        acc = __builtin_amdgcn_wmma_f32_16x16x32_f16(false, af, false, bf,
                                                     (short)0, acc, false, false);
    }

    store_tile(Out, ldo, row0, col0, M, Nn, lane, acc);
}

// =====================================================================
// h_new[b] = attn_b(53x53) @ val_b(53x64), K=53 padded to 64.
// Clamped-address loads + cndmask-zero on the K overrun (no exec branches).
// =====================================================================
__global__ void gemm_av(const float* __restrict__ A, long sA,   // attn, lda=53
                        const float* __restrict__ V, long sV,   // val rows, ldb=192
                        float* __restrict__ Out, long sO)       // h,   ldo=64
{
    int lane = threadIdx.x & 31;
    int z = blockIdx.z;
    const float* Ab = A + (long)z * sA;
    const float* Vb = V + (long)z * sV;
    float* Ob = Out + (long)z * sO;

    int row0 = blockIdx.y * 16;
    int col0 = blockIdx.x * 16;
    int mrow = row0 + (lane & 15); if (mrow > Cn - 1) mrow = Cn - 1;
    int ncol = col0 + (lane & 15);                     // < 64 always
    int kbA = (lane < 16) ? 0 : 16;
    int kbB = (lane < 16) ? 0 : 16;
    const float* arow = Ab + (long)mrow * Cn;

    v8f acc = {};
#pragma unroll
    for (int k0 = 0; k0 < 64; k0 += 32) {
        v16h af, bf;
#pragma unroll
        for (int i = 0; i < 8; ++i) {
            int kk = kbA + 2 * i;
            int k1 = k0 + kk;
            float x0 = arow[(k1     < Cn) ? k1     : Cn - 1];
            float x1 = arow[(k1 + 1 < Cn) ? k1 + 1 : Cn - 1];
            af[2 * i]     = (_Float16)((k1     < Cn) ? x0 : 0.f);
            af[2 * i + 1] = (_Float16)((k1 + 1 < Cn) ? x1 : 0.f);
        }
#pragma unroll
        for (int r = 0; r < 8; ++r) {
            int k1 = k0 + kbB + 2 * r;
            float x0 = Vb[(long)((k1     < Cn) ? k1     : Cn - 1) * G3n + ncol];
            float x1 = Vb[(long)((k1 + 1 < Cn) ? k1 + 1 : Cn - 1) * G3n + ncol];
            bf[2 * r]     = (_Float16)((k1     < Cn) ? x0 : 0.f);
            bf[2 * r + 1] = (_Float16)((k1 + 1 < Cn) ? x1 : 0.f);
        }
        acc = __builtin_amdgcn_wmma_f32_16x16x32_f16(false, af, false, bf,
                                                     (short)0, acc, false, false);
    }

    store_tile(Ob, Hn, row0, col0, Cn, Hn, lane, acc);
}

// =====================================================================
// gta_embed GEMM with fused xr = align * Amean:
//   e[r,n] = sum_k (align[r,k]*Amean[r/T,k]) * w_g1[n,k]
// M=9600 (exact), N=140 (clamped), K=2809: 87 unchecked k-tiles + remainder.
// Rows are only 4B-aligned (2809 odd) -> scalar b32 loads, but branch-free.
// b_g1 dropped: per-column shift cancels in training-mode BatchNorm.
// =====================================================================
__global__ void egemm(const float* __restrict__ alignp, const float* __restrict__ amean,
                      const float* __restrict__ wg1, float* __restrict__ ebuf)
{
    int lane = threadIdx.x & 31;
    int col0 = blockIdx.x * 16;
    int row0 = blockIdx.y * 16;

    int mrow = row0 + (lane & 15);             // < 9600 always
    int bb   = mrow / Tn;
    const float* arow = alignp + (long)mrow * C2n;
    const float* mp   = amean  + (long)bb   * C2n;
    int ncol = col0 + (lane & 15); if (ncol >= E1n) ncol = E1n - 1;
    const float* wrow = wg1 + (long)ncol * C2n;
    int kbA = (lane < 16) ? 0 : 16;
    int kbB = (lane < 16) ? 0 : 16;

    v8f acc = {};
    int k0 = 0;
    for (; k0 + 32 <= C2n; k0 += 32) {         // 87 full, unchecked tiles
        v16h af, bf;
#pragma unroll
        for (int i = 0; i < 8; ++i) {
            int kk = kbA + 2 * i;
            int k1 = k0 + kk;
            af[2 * i]     = (_Float16)(arow[k1]     * mp[k1]);
            af[2 * i + 1] = (_Float16)(arow[k1 + 1] * mp[k1 + 1]);
        }
#pragma unroll
        for (int r = 0; r < 8; ++r) {
            int k1 = k0 + kbB + 2 * r;
            bf[2 * r]     = (_Float16)wrow[k1];
            bf[2 * r + 1] = (_Float16)wrow[k1 + 1];
        }
        acc = __builtin_amdgcn_wmma_f32_16x16x32_f16(false, af, false, bf,
                                                     (short)0, acc, false, false);
    }
    {   // remainder tile (25 valid k), clamp + select
        v16h af, bf;
#pragma unroll
        for (int i = 0; i < 8; ++i) {
            int kk = kbA + 2 * i;
            int k1 = k0 + kk;
            int c0 = (k1     < C2n) ? k1     : C2n - 1;
            int c1 = (k1 + 1 < C2n) ? k1 + 1 : C2n - 1;
            float x0 = arow[c0] * mp[c0];
            float x1 = arow[c1] * mp[c1];
            af[2 * i]     = (_Float16)((k1     < C2n) ? x0 : 0.f);
            af[2 * i + 1] = (_Float16)((k1 + 1 < C2n) ? x1 : 0.f);
        }
#pragma unroll
        for (int r = 0; r < 8; ++r) {
            int k1 = k0 + kbB + 2 * r;
            float x0 = wrow[(k1     < C2n) ? k1     : C2n - 1];
            float x1 = wrow[(k1 + 1 < C2n) ? k1 + 1 : C2n - 1];
            bf[2 * r]     = (_Float16)((k1     < C2n) ? x0 : 0.f);
            bf[2 * r + 1] = (_Float16)((k1 + 1 < C2n) ? x1 : 0.f);
        }
        acc = __builtin_amdgcn_wmma_f32_16x16x32_f16(false, af, false, bf,
                                                     (short)0, acc, false, false);
    }

    store_tile(ebuf, E1n, row0, col0, BTn, E1n, lane, acc);
}

// ---------------- small helper kernels ----------------

// a_vec = w_ih @ emb_w ; c_vec = w_ih @ emb_b + b_ih   (rank-1 input projection)
__global__ void prep_vecs(const float* __restrict__ w_ih, const float* __restrict__ emb_w,
                          const float* __restrict__ emb_b, const float* __restrict__ b_ih,
                          float* __restrict__ avec, float* __restrict__ cvec)
{
    int j = blockIdx.x * blockDim.x + threadIdx.x;
    if (j >= G3n) return;
    float a = 0.f, c = 0.f;
    for (int e = 0; e < En; ++e) {
        a += w_ih[j * En + e] * emb_w[e];
        c += w_ih[j * En + e] * emb_b[e];
    }
    avec[j] = a;
    cvec[j] = c + b_ih[j];
}

// pack wq|wk|wv into one 192x64 weight block (single qkv GEMM per step)
__global__ void pack_qkvw(const float* __restrict__ wq, const float* __restrict__ wk,
                          const float* __restrict__ wv, float* __restrict__ wqkv)
{
    int idx = blockIdx.x * blockDim.x + threadIdx.x;
    if (idx >= G3n * Hn) return;
    int r = idx / Hn, c = idx % Hn;
    const float* src = (r < Hn) ? wq : ((r < 2 * Hn) ? wk : wv);
    int rr = (r < Hn) ? r : ((r < 2 * Hn) ? r - Hn : r - 2 * Hn);
    wqkv[idx] = src[rr * Hn + c];
}

__global__ void init_state(float* __restrict__ h, const float* __restrict__ u0,
                           float* __restrict__ u)
{
    int idx = blockIdx.x * blockDim.x + threadIdx.x;
    if (idx < NRn * Hn) h[idx] = 0.f;
    if (idx < Hn) u[idx] = u0[idx];
}

// GRU gate math: reads gh (=h@w_hh.T), x_t via rank-1 gi, writes hg.
__global__ void gru_gates(const float* __restrict__ x, int t,
                          const float* __restrict__ gh,
                          const float* __restrict__ avec, const float* __restrict__ cvec,
                          const float* __restrict__ b_hh,
                          const float* __restrict__ h, float* __restrict__ hg)
{
    int idx = blockIdx.x * blockDim.x + threadIdx.x;
    if (idx >= NRn * Hn) return;
    int n = idx / Hn, j = idx % Hn;
    int b = n / Cn, c = n % Cn;
    float xv = x[(long)b * Tn * Cn + (long)t * Cn + c];
    float ir = xv * avec[j]          + cvec[j];
    float iz = xv * avec[j +  Hn]    + cvec[j +  Hn];
    float in_= xv * avec[j + 2 * Hn] + cvec[j + 2 * Hn];
    const float* ghrow = gh + (long)n * G3n;
    float hr = ghrow[j]          + b_hh[j];
    float hz = ghrow[j +  Hn]    + b_hh[j +  Hn];
    float hn = ghrow[j + 2 * Hn] + b_hh[j + 2 * Hn];
    float r  = 1.f / (1.f + expf(-(ir + hr)));
    float zf = 1.f / (1.f + expf(-(iz + hz)));
    float nn = tanhf(in_ + r * hn);
    hg[idx] = (1.f - zf) * nn + zf * h[idx];
}

// 5-step power iteration on wv (64x64): updates u state, emits sigma.
__global__ void spectral_u(const float* __restrict__ wv, float* __restrict__ u,
                           float* __restrict__ sigma)
{
    __shared__ float red[64], us[64], vs[64];
    int j = threadIdx.x;
    us[j] = u[j];
    __syncthreads();
    for (int it = 0; it < 5; ++it) {
        float s = 0.f;                      // v = normalize(wv^T u)
        for (int i = 0; i < Hn; ++i) s += wv[i * Hn + j] * us[i];
        red[j] = s * s; __syncthreads();
        for (int off = 32; off; off >>= 1) { if (j < off) red[j] += red[j + off]; __syncthreads(); }
        float nv = sqrtf(red[0]) + 1e-12f;
        vs[j] = s / nv; __syncthreads();
        float su = 0.f;                     // u = normalize(wv v)
        for (int k = 0; k < Hn; ++k) su += wv[j * Hn + k] * vs[k];
        red[j] = su * su; __syncthreads();
        for (int off = 32; off; off >>= 1) { if (j < off) red[j] += red[j + off]; __syncthreads(); }
        float nu = sqrtf(red[0]) + 1e-12f;
        us[j] = su / nu; __syncthreads();
    }
    float wvv = 0.f;
    for (int k = 0; k < Hn; ++k) wvv += wv[j * Hn + k] * vs[k];
    red[j] = us[j] * wvv; __syncthreads();
    for (int off = 32; off; off >>= 1) { if (j < off) red[j] += red[j + off]; __syncthreads(); }
    if (j == 0) sigma[0] = red[0];
    u[j] = us[j];
}

// q += bq ; k += bk ; v = v/sigma + bv  (qkv layout: NRn x 192)
__global__ void qkv_post(float* __restrict__ qkv, const float* __restrict__ bq,
                         const float* __restrict__ bk, const float* __restrict__ bv,
                         const float* __restrict__ sigma)
{
    int idx = blockIdx.x * blockDim.x + threadIdx.x;
    if (idx >= NRn * G3n) return;
    int j = idx % G3n;
    float v = qkv[idx];
    if      (j <  Hn)    v += bq[j];
    else if (j < 2 * Hn) v += bk[j - Hn];
    else                 v = v / sigma[0] + bv[j - 2 * Hn];
    qkv[idx] = v;
}

// spectral radius per batch via power iteration on 53x53 scores.
__global__ void spec_radius(const float* __restrict__ S, long strideB,
                            float* __restrict__ m)
{
    const float* Sb = S + (long)blockIdx.x * strideB;
    __shared__ float xs[64], ys[64], red[64];
    int i = threadIdx.x;
    xs[i] = (i < Cn) ? 1.f : 0.f;
    __syncthreads();
    float lam = 1.f;
    for (int it = 0; it < 40; ++it) {
        float s = 0.f;
        if (i < Cn) for (int jj = 0; jj < Cn; ++jj) s += Sb[i * Cn + jj] * xs[jj];
        ys[i] = (i < Cn) ? s : 0.f;
        red[i] = ys[i] * ys[i]; __syncthreads();
        for (int off = 32; off; off >>= 1) { if (i < off) red[i] += red[i + off]; __syncthreads(); }
        float nrm = sqrtf(red[0]);
        lam = nrm;
        float inv = (nrm > 1e-20f) ? 1.f / nrm : 0.f;
        xs[i] = ys[i] * inv;
        __syncthreads();
    }
    if (i == 0) m[blockIdx.x] = fmaxf(lam, 1e-12f);
}

__global__ void scale_attn(float* __restrict__ S, long strideB, const float* __restrict__ m)
{
    int idx = blockIdx.x * blockDim.x + threadIdx.x;
    if (idx >= Bn * C2n) return;
    int b = idx / C2n, i = idx % C2n;
    S[(long)b * strideB + i] /= m[b];
}

__global__ void amean_k(const float* __restrict__ alignp, float* __restrict__ amean)
{
    int idx = blockIdx.x * blockDim.x + threadIdx.x;
    if (idx >= Bn * C2n) return;
    int b = idx / C2n, i = idx % C2n;
    const float* p = alignp + (long)b * Tn * C2n + i;
    float s = 0.f;
    for (int t = 0; t < Tn; ++t) s += p[(long)t * C2n];
    amean[idx] = s * (1.f / Tn);
}

// per-column mean/var over 9600 rows (training-mode BatchNorm stats)
__global__ void colstats(const float* __restrict__ e, float* __restrict__ mu,
                         float* __restrict__ var)
{
    int col = blockIdx.x;
    __shared__ float s1[256], s2[256];
    float a = 0.f, b2 = 0.f;
    for (int r = threadIdx.x; r < BTn; r += 256) {
        float v = e[(long)r * E1n + col];
        a += v; b2 += v * v;
    }
    s1[threadIdx.x] = a; s2[threadIdx.x] = b2; __syncthreads();
    for (int off = 128; off; off >>= 1) {
        if (threadIdx.x < off) { s1[threadIdx.x] += s1[threadIdx.x + off];
                                 s2[threadIdx.x] += s2[threadIdx.x + off]; }
        __syncthreads();
    }
    if (threadIdx.x == 0) {
        float mm = s1[0] / (float)BTn;
        mu[col]  = mm;
        var[col] = s2[0] / (float)BTn - mm * mm;
    }
}

__global__ void bnrelu(float* __restrict__ e, const float* __restrict__ mu,
                       const float* __restrict__ var, const float* __restrict__ g,
                       const float* __restrict__ bb)
{
    long idx = (long)blockIdx.x * blockDim.x + threadIdx.x;
    if (idx >= (long)BTn * E1n) return;
    int col = (int)(idx % E1n);
    float v = (e[idx] - mu[col]) * rsqrtf(var[col] + 1e-5f) * g[col] + bb[col];
    e[idx] = fmaxf(v, 0.f);
}

// fused: a[r] = relu(e_row @ w_g2^T + b_g2) @ w_g3^T + b_g3 (no E2-sized buffer)
__global__ void attend(const float* __restrict__ e, const float* __restrict__ wg2,
                       const float* __restrict__ bg2, const float* __restrict__ wg3,
                       const float* __restrict__ bg3, float* __restrict__ a)
{
    __shared__ float er[E1n];
    __shared__ float red[256];
    int r = blockIdx.x;
    for (int k = threadIdx.x; k < E1n; k += 256) er[k] = e[(long)r * E1n + k];
    __syncthreads();
    float acc = 0.f;
    for (int j = threadIdx.x; j < E2n; j += 256) {
        float s = bg2[j];
        const float* wrow = wg2 + (long)j * E1n;
        for (int k = 0; k < E1n; ++k) s += wrow[k] * er[k];
        acc += fmaxf(s, 0.f) * wg3[j];
    }
    red[threadIdx.x] = acc; __syncthreads();
    for (int off = 128; off; off >>= 1) {
        if (threadIdx.x < off) red[threadIdx.x] += red[threadIdx.x + off];
        __syncthreads();
    }
    if (threadIdx.x == 0) a[r] = red[0] + bg3[0];
}

__global__ void softmaxT(float* __restrict__ a)
{
    int b = blockIdx.x;
    __shared__ float red[256];
    int t = threadIdx.x;
    float v = (t < Tn) ? a[b * Tn + t] : -1e30f;
    red[t] = v; __syncthreads();
    for (int off = 128; off; off >>= 1) {
        if (t < off) red[t] = fmaxf(red[t], red[t + off]);
        __syncthreads();
    }
    float mx = red[0]; __syncthreads();
    float ex = (t < Tn) ? expf(v - mx) : 0.f;
    red[t] = ex; __syncthreads();
    for (int off = 128; off; off >>= 1) {
        if (t < off) red[t] += red[t + off];
        __syncthreads();
    }
    float s = red[0];
    if (t < Tn) a[b * Tn + t] = ex / s;
}

__global__ void dnc_k(const float* __restrict__ alignp, const float* __restrict__ a,
                      float* __restrict__ dnc)
{
    int idx = blockIdx.x * blockDim.x + threadIdx.x;
    if (idx >= Bn * C2n) return;
    int b = idx / C2n, i = idx % C2n;
    const float* p = alignp + (long)b * Tn * C2n + i;
    float s = 0.f;
    for (int t = 0; t < Tn; ++t) s += p[(long)t * C2n] * a[b * Tn + t];
    dnc[idx] = s;
}

__global__ void logits_k(const float* __restrict__ dnc, const float* __restrict__ w_clf,
                         const float* __restrict__ b_clf, float* __restrict__ logits)
{
    int idx = threadIdx.x;              // 96 = B*O
    if (idx >= Bn * 2) return;
    int b = idx / 2, o = idx % 2;
    float s = b_clf[o];
    const float* drow = dnc + (long)b * C2n;
    const float* wrow = w_clf + (long)o * C2n;
    for (int i = 0; i < C2n; ++i) s += drow[i] * wrow[i];
    logits[idx] = s;
}

// =====================================================================
extern "C" void kernel_launch(void* const* d_in, const int* in_sizes, int n_in,
                              void* d_out, int out_size, void* d_ws, size_t ws_size,
                              hipStream_t stream)
{
    (void)in_sizes; (void)n_in; (void)out_size; (void)ws_size;

    const float* x     = (const float*)d_in[0];
    const float* emb_w = (const float*)d_in[1];
    const float* emb_b = (const float*)d_in[2];
    const float* w_ih  = (const float*)d_in[3];
    const float* w_hh  = (const float*)d_in[4];
    const float* b_ih  = (const float*)d_in[5];
    const float* b_hh  = (const float*)d_in[6];
    const float* wq    = (const float*)d_in[7];
    const float* bq    = (const float*)d_in[8];
    const float* wk    = (const float*)d_in[9];
    const float* bk    = (const float*)d_in[10];
    const float* wv    = (const float*)d_in[11];
    const float* bv    = (const float*)d_in[12];
    const float* u0    = (const float*)d_in[13];
    const float* w_g1  = (const float*)d_in[14];
    /* b_g1 (d_in[15]) intentionally unused: cancels under BatchNorm */
    const float* bn_g  = (const float*)d_in[16];
    const float* bn_b  = (const float*)d_in[17];
    const float* w_g2  = (const float*)d_in[18];
    const float* b_g2  = (const float*)d_in[19];
    const float* w_g3  = (const float*)d_in[20];
    const float* b_g3  = (const float*)d_in[21];
    const float* w_clf = (const float*)d_in[22];
    const float* b_clf = (const float*)d_in[23];

    // ---- outputs: logits(96) | dnc(48*2809) | align(48*200*2809) ----
    float* out     = (float*)d_out;
    float* logits  = out;
    float* dnc     = out + Bn * 2;
    float* alignp  = out + Bn * 2 + Bn * C2n;

    // ---- workspace carve (≈11.3 MB of f32) ----
    float* ws   = (float*)d_ws;
    float* h    = ws;  ws += NRn * Hn;
    float* hg   = ws;  ws += NRn * Hn;
    float* ghb  = ws;  ws += NRn * G3n;
    float* qkv  = ws;  ws += NRn * G3n;
    float* wqkv = ws;  ws += G3n * Hn;
    float* avec = ws;  ws += G3n;
    float* cvec = ws;  ws += G3n;
    float* u    = ws;  ws += Hn;
    float* sig  = ws;  ws += 4;
    float* mrad = ws;  ws += Bn;
    float* amean= ws;  ws += Bn * C2n;
    float* ebuf = ws;  ws += (long)BTn * E1n;
    float* mu   = ws;  ws += E1n;
    float* var  = ws;  ws += E1n;
    float* abuf = ws;  ws += BTn;

    const int THR = 256;
    prep_vecs<<<1, 256, 0, stream>>>(w_ih, emb_w, emb_b, b_ih, avec, cvec);
    pack_qkvw<<<(G3n * Hn + THR - 1) / THR, THR, 0, stream>>>(wq, wk, wv, wqkv);
    init_state<<<(NRn * Hn + THR - 1) / THR, THR, 0, stream>>>(h, u0, u);

    const long sQKV = (long)Cn * G3n;   // batch stride inside qkv
    const long sALN = (long)Tn * C2n;   // batch stride inside align
    const long sH   = (long)Cn * Hn;    // batch stride inside h

    for (int t = 0; t < Tn; ++t) {
        // gh = h @ w_hh^T                    (2544 x 192, K=64)
        gemm_k64<<<dim3(G3n / 16, NRn / 16, 1), 32, 0, stream>>>(
            h, Hn, 0L, w_hh, Hn, 0L, ghb, G3n, 0L, NRn, G3n);
        // GRU gates -> hg
        gru_gates<<<(NRn * Hn + THR - 1) / THR, THR, 0, stream>>>(
            x, t, ghb, avec, cvec, b_hh, h, hg);
        // spectral norm power iteration on wv (updates u, emits sigma)
        spectral_u<<<1, 64, 0, stream>>>(wv, u, sig);
        // q|k|v = hg @ wqkv^T                (2544 x 192, K=64, one launch)
        gemm_k64<<<dim3(G3n / 16, NRn / 16, 1), 32, 0, stream>>>(
            hg, Hn, 0L, wqkv, Hn, 0L, qkv, G3n, 0L, NRn, G3n);
        qkv_post<<<(NRn * G3n + THR - 1) / THR, THR, 0, stream>>>(qkv, bq, bk, bv, sig);
        // scores[b] = q_b @ k_b^T -> written straight into align[:, t]
        gemm_k64<<<dim3(4, 4, Bn), 32, 0, stream>>>(
            qkv, G3n, sQKV, qkv + Hn, G3n, sQKV,
            alignp + (long)t * C2n, Cn, sALN, Cn, Cn);
        // spectral-radius normalization (in place -> final attn)
        spec_radius<<<Bn, 64, 0, stream>>>(alignp + (long)t * C2n, sALN, mrad);
        scale_attn<<<(Bn * C2n + THR - 1) / THR, THR, 0, stream>>>(
            alignp + (long)t * C2n, sALN, mrad);
        // h_new[b] = attn_b @ val_b          (53 x 64, K=53)
        gemm_av<<<dim3(4, 4, Bn), 32, 0, stream>>>(
            alignp + (long)t * C2n, sALN, qkv + 2 * Hn, sQKV, h, sH);
    }

    // ---- gta_attention over time ----
    amean_k<<<(Bn * C2n + THR - 1) / THR, THR, 0, stream>>>(alignp, amean);
    egemm<<<dim3((E1n + 15) / 16, BTn / 16), 32, 0, stream>>>(alignp, amean, w_g1, ebuf);
    colstats<<<E1n, 256, 0, stream>>>(ebuf, mu, var);
    bnrelu<<<(int)(((long)BTn * E1n + THR - 1) / THR), THR, 0, stream>>>(
        ebuf, mu, var, bn_g, bn_b);
    attend<<<BTn, 256, 0, stream>>>(ebuf, w_g2, b_g2, w_g3, b_g3, abuf);
    softmaxT<<<Bn, 256, 0, stream>>>(abuf);
    dnc_k<<<(Bn * C2n + THR - 1) / THR, THR, 0, stream>>>(alignp, abuf, dnc);
    logits_k<<<1, 96, 0, stream>>>(dnc, w_clf, b_clf, logits);
}
